// ST_Block_32512902431542
// MI455X (gfx1250) — matrix operand order repaired and checked by reading, not compiled, not run on hardware
//
#include <hip/hip_runtime.h>

typedef __attribute__((ext_vector_type(16))) __bf16 v16bf;
typedef __attribute__((ext_vector_type(8)))  __bf16 v8bf;
typedef __attribute__((ext_vector_type(8)))  float  v8f;
typedef __attribute__((ext_vector_type(4)))  float  v4f;
typedef __attribute__((__vector_size__(16))) int    vi4;

#define SHUF16(lo, hi) __builtin_shufflevector(lo, hi, 0,1,2,3,4,5,6,7,8,9,10,11,12,13,14,15)

#define NN     2000
#define NPAD   2016
#define NC     3072
#define NPTS   48000   // 2000*24 per batch

// ---------------------------------------------------------------------------
// CDNA5 async global->LDS copy (ASYNCcnt path), with portable fallback
// ---------------------------------------------------------------------------
#if defined(__has_builtin)
#if __has_builtin(__builtin_amdgcn_global_load_async_to_lds_b128)
#define ASYNC_COPY_OK 1
#endif
#endif

__device__ __forceinline__ void async_cp16(const __bf16* g, __bf16* l)
{
#ifdef ASYNC_COPY_OK
    __builtin_amdgcn_global_load_async_to_lds_b128(
        (__attribute__((address_space(1))) vi4*)g,
        (__attribute__((address_space(3))) vi4*)l, 0, 0);
#else
    *(v8bf*)l = *(const v8bf*)g;   // global_load_b128 + ds_store_b128
#endif
}

__device__ __forceinline__ void async_wait_all()
{
#ifdef ASYNC_COPY_OK
#if __has_builtin(__builtin_amdgcn_s_wait_asynccnt)
    __builtin_amdgcn_s_wait_asynccnt(0);
#else
    asm volatile("s_wait_asynccnt 0" ::: "memory");
#endif
#endif
}

// ---------------------------------------------------------------------------
// Prep: theta, tp row-sum polynomials, per-(k,b,t) Horner scalars G[11][4][24]
// ---------------------------------------------------------------------------
__global__ void prep_kernel(const float* __restrict__ tp, const float* __restrict__ STE,
                            const float* __restrict__ w1, const float* __restrict__ b1,
                            const float* __restrict__ w2, const float* __restrict__ b2,
                            float* __restrict__ G, float* __restrict__ theta_out)
{
    __shared__ float h1[200];        // [4][5][10]
    __shared__ float th[200];        // theta [4][5][10]
    __shared__ float v[6][24];       // tp^k @ ones
    __shared__ float tps[5][24];     // row sums of tp_list
    __shared__ float wsh[960];       // w [4][10][24]
    __shared__ float dsh[10][11];    // spatial Bernstein -> monomial coeffs
    __shared__ float esh[5][6];      // temporal
    const int tid = threadIdx.x;

    if (tid == 0) {
        int Cb[11][11];
        for (int i = 0; i < 11; ++i) {
            Cb[i][0] = 1;
            for (int j = 1; j <= i; ++j)
                Cb[i][j] = Cb[i-1][j-1] + ((j <= i-1) ? Cb[i-1][j] : 0);
            for (int j = i+1; j < 11; ++j) Cb[i][j] = 0;
        }
        for (int o = 0; o < 10; ++o)
            for (int k = 0; k <= 10; ++k) {
                int s = 0;
                for (int j = 0; j <= o && j <= k; ++j) {
                    int a = k - j;
                    if (a <= 10 - o) s += Cb[10-o][a] * Cb[o][j] * ((j & 1) ? -1 : 1);
                }
                dsh[o][k] = (float)Cb[10][o] * (float)s / 1024.0f;
            }
        for (int t = 0; t < 5; ++t)
            for (int k = 0; k <= 5; ++k) {
                int s = 0;
                for (int j = 0; j <= t && j <= k; ++j) {
                    int a = k - j;
                    if (a <= 5 - t) s += Cb[5-t][a] * Cb[t][j] * ((j & 1) ? -1 : 1);
                }
                esh[t][k] = (float)Cb[5][t] * (float)s / 32.0f;
            }
    }
    __syncthreads();

    if (tid < 200) {                 // h1[b,t,s] = w1 @ STE + b1
        int b = tid / 50, r = tid % 50, t = r / 10, s = r % 10;
        float a = b1[t];
        for (int i = 0; i < 5; ++i) a += w1[t*5 + i] * STE[b*50 + i*10 + s];
        h1[tid] = a;
    }
    __syncthreads();
    if (tid < 200) {                 // theta[b,t,s] = relu(w2 @ h1 + b2)
        int b = tid / 50, r = tid % 50, t = r / 10, s = r % 10;
        float a = b2[s];
        for (int i = 0; i < 10; ++i) a += w2[s*10 + i] * h1[b*50 + t*10 + i];
        a = fmaxf(a, 0.0f);
        th[tid] = a;
        theta_out[tid] = a;          // second output
    }
    if (tid < 24) v[0][tid] = 1.0f;
    __syncthreads();
    for (int k = 1; k <= 5; ++k) {   // v[k] = tp @ v[k-1]
        if (tid < 24) {
            float a = 0.0f;
            for (int j = 0; j < 24; ++j) a += tp[tid*24 + j] * v[k-1][j];
            v[k][tid] = a;
        }
        __syncthreads();
    }
    if (tid < 120) {                 // tps[t][n] = sum_k e[t][k] v[k][n]
        int t = tid / 24, n = tid % 24;
        float a = 0.0f;
        for (int k = 0; k <= 5; ++k) a += esh[t][k] * v[k][n];
        tps[t][n] = a;
    }
    __syncthreads();
    for (int idx = tid; idx < 960; idx += 256) {   // w[b,s,n] = sum_t theta[b,t,s] tps[t,n]
        int b = idx / 240, r = idx % 240, s = r / 24, n = r % 24;
        float a = 0.0f;
        for (int t = 0; t < 5; ++t) a += th[b*50 + t*10 + s] * tps[t][n];
        wsh[idx] = a;
    }
    __syncthreads();
    for (int idx = tid; idx < 1056; idx += 256) {  // G[k][b][t] = sum_o d[o][k] w[b,o,t]
        int k = idx / 96, b = (idx % 96) / 24, t = idx % 24;
        float a = 0.0f;
        for (int o = 0; o < 10; ++o) a += dsh[o][k] * wsh[b*240 + o*24 + t];
        G[idx] = a;
    }
}

// ---------------------------------------------------------------------------
// Pack sp_matrix -> bf16, zero-padded K (columns 2000..2015)
// ---------------------------------------------------------------------------
__global__ void pack_M(const float* __restrict__ sp, __bf16* __restrict__ Mbf)
{
    int idx = blockIdx.x * 256 + threadIdx.x;        // 2000*2016 = 4,032,000
    if (idx >= NN * NPAD) return;
    int m = idx / NPAD, n = idx % NPAD;
    float v = (n < NN) ? sp[m * NN + n] : 0.0f;
    Mbf[idx] = (__bf16)v;
}

// ---------------------------------------------------------------------------
// Build Xc (fp32, transposed [col][n], zero pad) and R0 (bf16) = x * g_10
// ---------------------------------------------------------------------------
__global__ void pack_X(const float* __restrict__ x, const float* __restrict__ G,
                       float* __restrict__ Xc, __bf16* __restrict__ R0)
{
    int idx = blockIdx.x * 256 + threadIdx.x;        // 3072*2016 = 6,193,152
    if (idx >= NC * NPAD) return;
    int c = idx / NPAD, m = idx % NPAD;
    int b = c / 768, t = c % 24, f = (c / 24) & 31;
    float xv = 0.0f;
    if (m < NN) xv = x[(((b*32 + f) * NN) + m) * 24 + t];
    Xc[idx] = xv;
    R0[idx] = (__bf16)(xv * G[10*96 + b*24 + t]);
}

// ---------------------------------------------------------------------------
// Horner GEMM step:  Y[m,c] = sum_n M[m,n] R[n,c] + Xc[m,c] * g_k[col(c)]
//   A (M) staged through LDS via async copies, double-buffered; B direct.
//   Block = 8 waves, all sharing one 80-row strip; wave tile 80x32 (5x2 frags)
// ---------------------------------------------------------------------------
__global__ void __launch_bounds__(256) horner_gemm(
    const __bf16* __restrict__ Mbf, const __bf16* __restrict__ Rin,
    const float*  __restrict__ Xc,  const float*  __restrict__ G,
    __bf16* __restrict__ Rout, float* __restrict__ XstT,
    int kidx, int is_final)
{
    __shared__ __bf16 Atile[2][80][40];   // 80 rows x 32 K (+8 pad), 2 buffers

    const int tid   = threadIdx.x;
    const int lane  = tid & 31;
    const int wave  = tid >> 5;
    const int m0    = (blockIdx.x / 12) * 80;                 // 25 row strips, 12 blocks each
    const int c0    = ((blockIdx.x % 12) * 8 + wave) * 32;    // 96 col strips
    const int lhalf = lane >> 4;
    const int l15   = lane & 15;

    // A-tile copy assignment: 320 x 16B chunks; chunk = row*4 + quarter
    const int ra = tid >> 2, qa = tid & 3;

    v8f acc[5][2] = {};

    const __bf16* Bbase = Rin + (size_t)(c0 + l15) * NPAD + lhalf * 16;

    auto load_tile = [&](int buf, int kk) {
        async_cp16(Mbf + (size_t)(m0 + ra) * NPAD + kk + qa * 8,
                   &Atile[buf][ra][qa * 8]);
        if (tid < 64)
            async_cp16(Mbf + (size_t)(m0 + 64 + ra) * NPAD + kk + qa * 8,
                       &Atile[buf][64 + ra][qa * 8]);
    };

    load_tile(0, 0);
    async_wait_all();
    __syncthreads();

    for (int kt = 0; kt < 63; ++kt) {
        const int kk  = kt * 32;
        const int buf = kt & 1;
        if (kt + 1 < 63) load_tile(buf ^ 1, kk + 32);

        v16bf a[5], b[2];
#pragma unroll
        for (int j = 0; j < 2; ++j) {     // B: lane=col, contiguous 16 K values
            const __bf16* q = Bbase + (size_t)j * 16 * NPAD + kk;
            v8bf lo = *(const v8bf*)q;
            v8bf hi = *(const v8bf*)(q + 8);
            b[j] = SHUF16(lo, hi);
        }
#pragma unroll
        for (int i = 0; i < 5; ++i) {     // A from LDS: lanes<16 K{0..15}, >=16 K{16..31}
            const __bf16* ar = &Atile[buf][i * 16 + l15][lhalf * 16];
            v8bf lo = *(const v8bf*)ar;
            v8bf hi = *(const v8bf*)(ar + 8);
            a[i] = SHUF16(lo, hi);
        }
#pragma unroll
        for (int i = 0; i < 5; ++i)
#pragma unroll
            for (int j = 0; j < 2; ++j)
                acc[i][j] = __builtin_amdgcn_wmma_f32_16x16x32_bf16(
                    false, a[i], false, b[j], (short)0, acc[i][j], false, false);

        if (kt + 1 < 63) { async_wait_all(); __syncthreads(); }
    }

    // Epilogue: += Xc * g_k ; store bf16 (next R) or fp32 (final x_st)
#pragma unroll
    for (int j = 0; j < 2; ++j) {
        int c  = c0 + j * 16 + l15;
        int bb = c / 768, tt = c % 24;
        float g = G[kidx * 96 + bb * 24 + tt];
        size_t colbase = (size_t)c * NPAD;
#pragma unroll
        for (int i = 0; i < 5; ++i) {
            int mb = m0 + i * 16 + lhalf * 8;     // 8 consecutive rows per lane
            v4f x0 = *(const v4f*)(Xc + colbase + mb);
            v4f x1 = *(const v4f*)(Xc + colbase + mb + 4);
            float outv[8];
#pragma unroll
            for (int r = 0; r < 8; ++r) {
                float xr = (r < 4) ? x0[r] : x1[r - 4];
                outv[r] = acc[i][j][r] + xr * g;
            }
            if (is_final) {
                v4f o0, o1;
#pragma unroll
                for (int r = 0; r < 4; ++r) { o0[r] = outv[r]; o1[r] = outv[r + 4]; }
                *(v4f*)(XstT + colbase + mb)     = o0;
                *(v4f*)(XstT + colbase + mb + 4) = o1;
            } else {
                v8bf ob;
#pragma unroll
                for (int r = 0; r < 8; ++r) ob[r] = (__bf16)outv[r];
                *(v8bf*)(Rout + colbase + mb) = ob;
            }
        }
    }
}

// ---------------------------------------------------------------------------
// 1x1 conv over cat(x, x_st) + BatchNorm (eval). W (64x64) in LDS.
// ---------------------------------------------------------------------------
__global__ void __launch_bounds__(256) mlp_bn(
    const float* __restrict__ Xc, const float* __restrict__ XstT,
    const float* __restrict__ W,  const float* __restrict__ Wb,
    const float* __restrict__ gamma, const float* __restrict__ beta,
    const float* __restrict__ mean,  const float* __restrict__ var,
    float* __restrict__ out)
{
    __shared__ float Wl[64 * 64];
    __shared__ float scl[64], sft[64];
    const int tid = threadIdx.x;
    for (int i = tid; i < 4096; i += 256) Wl[i] = W[i];
    if (tid < 64) {
        float s = gamma[tid] * rsqrtf(var[tid] + 1e-5f);
        scl[tid] = s;
        sft[tid] = (Wb[tid] - mean[tid]) * s + beta[tid];
    }
    __syncthreads();

    const int b = blockIdx.y;
    const int p = blockIdx.x * 256 + tid;      // p = n*24 + t
    if (p >= NPTS) return;
    const int n = p / 24, t = p % 24;

    float xa[32], xs[32];
#pragma unroll
    for (int f = 0; f < 32; ++f) {
        size_t off = (size_t)((b*32 + f) * 24 + t) * NPAD + n;
        xa[f] = Xc[off];
        xs[f] = XstT[off];
    }
    for (int o = 0; o < 64; ++o) {
        float a = 0.0f;
#pragma unroll
        for (int f = 0; f < 32; ++f)
            a += Wl[o*64 + f] * xa[f] + Wl[o*64 + 32 + f] * xs[f];
        out[(size_t)(b*64 + o) * NPTS + p] = a * scl[o] + sft[o];
    }
}

// ---------------------------------------------------------------------------
// Launch
// ---------------------------------------------------------------------------
extern "C" void kernel_launch(void* const* d_in, const int* in_sizes, int n_in,
                              void* d_out, int out_size, void* d_ws, size_t ws_size,
                              hipStream_t stream)
{
    const float* x   = (const float*)d_in[0];
    const float* sp  = (const float*)d_in[1];
    const float* tp  = (const float*)d_in[2];
    const float* STE = (const float*)d_in[3];
    const float* w1  = (const float*)d_in[4];
    const float* b1  = (const float*)d_in[5];
    const float* w2  = (const float*)d_in[6];
    const float* b2  = (const float*)d_in[7];
    const float* mw  = (const float*)d_in[8];
    const float* mb  = (const float*)d_in[9];
    const float* bg  = (const float*)d_in[10];
    const float* bb  = (const float*)d_in[11];
    const float* bm  = (const float*)d_in[12];
    const float* bv  = (const float*)d_in[13];
    float* out = (float*)d_out;

    char* ws = (char*)d_ws;
    float*  G    = (float*)(ws + 0);          //  4,224 B
    __bf16* Mbf  = (__bf16*)(ws + 8192);      //  8,064,000 B  (2000x2016 bf16)
    float*  Xc   = (float*)(ws + 8072192);    // 24,772,608 B  (3072x2016 f32)
    float*  Xst  = (float*)(ws + 32844800);   // 24,772,608 B
    __bf16* R0   = (__bf16*)(ws + 57617408);  // 12,386,304 B
    __bf16* R1   = (__bf16*)(ws + 70003712);  // 12,386,304 B  (total ~82.4 MB)

    prep_kernel<<<1, 256, 0, stream>>>(tp, STE, w1, b1, w2, b2, G, out + 12288000);
    pack_M<<<(NN * NPAD) / 256, 256, 0, stream>>>(sp, Mbf);
    pack_X<<<(NC * NPAD) / 256, 256, 0, stream>>>(x, G, Xc, R0);

    __bf16* rin = R0;
    __bf16* rout = R1;
    for (int k = 9; k >= 0; --k) {
        horner_gemm<<<300, 256, 0, stream>>>(Mbf, rin, Xc, G, rout, Xst, k, (k == 0) ? 1 : 0);
        __bf16* tswap = rin; rin = rout; rout = tswap;
    }

    mlp_bn<<<dim3((NPTS + 255) / 256, 4), 256, 0, stream>>>(
        Xc, Xst, mw, mb, bg, bb, bm, bv, out);
}
